// MSC_32409823216289
// MI455X (gfx1250) — matrix operand order repaired and checked by reading, not compiled, not run on hardware
//
#include <hip/hip_runtime.h>
#include <cstdint>
#include <cstddef>

// ---------------------------------------------------------------------------
// Problem constants (match reference)
// ---------------------------------------------------------------------------
static constexpr int Bsz  = 4;
static constexpr int NX   = 1024;
static constexpr int NY   = 1024;
static constexpr int Cdim = 768;
static constexpr int NH   = 12;
static constexpr int Dh   = 64;          // head dim
static constexpr float LN_EPS = 1e-5f;

typedef __bf16 bf16_t;
typedef __attribute__((ext_vector_type(16))) __bf16 v16bf;
typedef __attribute__((ext_vector_type(8)))  float  v8f;

// ---------------------------------------------------------------------------
// gfx1250 async global->LDS copy (ASYNCcnt-tracked). The builtin's parameter
// types (from the compiler diagnostic) are pointers to a 16-byte int vector:
//   arg0: addrspace(1) v4i*  (global source)
//   arg1: addrspace(3) v4i*  (LDS destination)
//   arg2: imm offset, arg3: imm cpol
// Guarded: falls back to plain 16B copies if the builtin is not declared.
// ---------------------------------------------------------------------------
#if defined(__has_builtin)
#  if __has_builtin(__builtin_amdgcn_global_load_async_to_lds_b128)
#    define HAVE_ASYNC_G2L 1
#  endif
#endif

typedef int async_v4i __attribute__((__vector_size__(4 * sizeof(int))));
typedef __attribute__((address_space(1))) async_v4i* gptr_v4i;
typedef __attribute__((address_space(3))) async_v4i* lptr_v4i;

__device__ __forceinline__ void async_g2l_b128(const void* g, void* l) {
#ifdef HAVE_ASYNC_G2L
  __builtin_amdgcn_global_load_async_to_lds_b128(
      (gptr_v4i)(uintptr_t)g, (lptr_v4i)(uintptr_t)l, 0, 0);
#else
  *(uint4*)l = *(const uint4*)g;
#endif
}

// Allow up to 32 async copies still in flight (retires the earlier Q chunks).
__device__ __forceinline__ void wait_async_le32() {
#ifdef HAVE_ASYNC_G2L
#  if __has_builtin(__builtin_amdgcn_s_wait_asynccnt)
  __builtin_amdgcn_s_wait_asynccnt(32);
#  else
  asm volatile("s_wait_asynccnt 32" ::: "memory");
#  endif
#endif
}

__device__ __forceinline__ void wait_async_0() {
#ifdef HAVE_ASYNC_G2L
#  if __has_builtin(__builtin_amdgcn_s_wait_asynccnt)
  __builtin_amdgcn_s_wait_asynccnt(0);
#  else
  asm volatile("s_wait_asynccnt 0" ::: "memory");
#  endif
#endif
}

// ---------------------------------------------------------------------------
// WMMA helpers — CDNA5 V_WMMA_F32_16X16X32_BF16 (D = A(16x32) * B(32x16) + C)
// Fragment layouts loaded by these helpers (wave32; contiguous k, not the ISA table order):
//   A 16-bit 16x32 : lanes 0-15 row M=lane, K=0..15; lanes 16-31 same
//                    rows, K=16..31  (2 halfs per VGPR)
//   B 16-bit 32x16 : lanes 0-15 col N=lane, K=0..15; lanes 16-31 K=16..31 (same k order as A)
//   C/D f32 16x16  : VGPR r -> M = r + 8*(lane>=16), N = lane&15
// ---------------------------------------------------------------------------
__device__ __forceinline__ v8f wmma_bf16(v16bf a, v16bf b, v8f c) {
  return __builtin_amdgcn_wmma_f32_16x16x32_bf16(
      /*neg_a=*/false, a, /*neg_b=*/false, b,
      /*c_mod=*/(short)0, c, /*reuse_a=*/false, /*reuse_b=*/false);
}

// A fragment: matrix stored row-major [rows, rstride], tile origin (row0,col0)
__device__ __forceinline__ v16bf load_a_frag(const bf16_t* base, int rstride,
                                             int row0, int col0) {
  const int lane = threadIdx.x & 31;
  const int r    = row0 + (lane & 15);
  const int klo  = (lane >> 4) * 16;
  const bf16_t* p = base + (size_t)r * rstride + col0;
  v16bf f;
#pragma unroll
  for (int j = 0; j < 8; ++j) f[j]     = p[klo + j];
#pragma unroll
  for (int j = 0; j < 8; ++j) f[8 + j] = p[klo + 8 + j];
  return f;
}

// B fragment where B[kk][n] = W[n][kk]  (W row-major [N, stride], K contiguous)
// -> out = A @ W^T. Each lane reads 16 contiguous halfs (vectorizes to b128).
__device__ __forceinline__ v16bf load_bT_frag(const bf16_t* W, int stride,
                                              int n0, int k0) {
  const int lane = threadIdx.x & 31;
  const int n    = n0 + (lane & 15);
  const int kb   = k0 + ((lane >> 4) << 4);
  const bf16_t* p = W + (size_t)n * stride + kb;
  v16bf f;
#pragma unroll
  for (int j = 0; j < 16; ++j) f[j] = p[j];
  return f;
}

// B fragment where B[kk][n] = V[kk][n]  (V row-major [K, stride]) — strided,
// only used against LDS-staged tiles.
__device__ __forceinline__ v16bf load_b_rm_frag(const bf16_t* V, int stride,
                                                int k0, int n0) {
  const int lane = threadIdx.x & 31;
  const int n    = n0 + (lane & 15);
  const int kb   = k0 + ((lane >> 4) << 4);
  v16bf f;
#pragma unroll
  for (int j = 0; j < 16; ++j) f[j] = V[(size_t)(kb + j) * stride + n];
  return f;
}

// ---------------------------------------------------------------------------
// fp32 -> bf16 cast
// ---------------------------------------------------------------------------
__global__ void cast_f32_bf16(const float* __restrict__ s,
                              bf16_t* __restrict__ d, int n) {
  int i = blockIdx.x * blockDim.x + threadIdx.x;
  if (i < n) d[i] = (bf16_t)s[i];
}

// ---------------------------------------------------------------------------
// Fused multi-scale depthwise conv (K=3,5,7, same pad) + LayerNorm -> bf16.
// One block per (b, n) row; 256 threads (8 waves), block-reduced LN stats.
// ---------------------------------------------------------------------------
__global__ void conv_ln_kernel(const float* __restrict__ y,
                               const float* __restrict__ w1, const float* __restrict__ b1,
                               const float* __restrict__ w2, const float* __restrict__ b2,
                               const float* __restrict__ w3, const float* __restrict__ b3,
                               const float* __restrict__ lnw, const float* __restrict__ lnb,
                               bf16_t* __restrict__ ynorm) {
  const int b = blockIdx.x / NY;
  const int n = blockIdx.x % NY;
  __shared__ float rowbuf[Cdim];
  __shared__ float redsum[8], redsq[8];

  const int lane = threadIdx.x & 31;
  const int wave = threadIdx.x >> 5;

  float lsum = 0.f, lsq = 0.f;
  for (int c = threadIdx.x; c < Cdim; c += blockDim.x) {
    float s = b1[c] + b2[c] + b3[c];
#pragma unroll
    for (int t = 0; t < 3; ++t) {
      int idx = n - 1 + t;
      if (idx >= 0 && idx < NY)
        s += w1[c * 3 + t] * y[((size_t)b * NY + idx) * Cdim + c];
    }
#pragma unroll
    for (int t = 0; t < 5; ++t) {
      int idx = n - 2 + t;
      if (idx >= 0 && idx < NY)
        s += w2[c * 5 + t] * y[((size_t)b * NY + idx) * Cdim + c];
    }
#pragma unroll
    for (int t = 0; t < 7; ++t) {
      int idx = n - 3 + t;
      if (idx >= 0 && idx < NY)
        s += w3[c * 7 + t] * y[((size_t)b * NY + idx) * Cdim + c];
    }
    rowbuf[c] = s;
    lsum += s;
    lsq  += s * s;
  }
#pragma unroll
  for (int off = 16; off; off >>= 1) {
    lsum += __shfl_xor(lsum, off, 32);
    lsq  += __shfl_xor(lsq,  off, 32);
  }
  if (lane == 0) { redsum[wave] = lsum; redsq[wave] = lsq; }
  __syncthreads();
  float tsum = 0.f, tsq = 0.f;
#pragma unroll
  for (int i = 0; i < 8; ++i) { tsum += redsum[i]; tsq += redsq[i]; }
  const float mu  = tsum / (float)Cdim;
  const float var = tsq / (float)Cdim - mu * mu;
  const float inv = rsqrtf(var + LN_EPS);
  for (int c = threadIdx.x; c < Cdim; c += blockDim.x) {
    float v = (rowbuf[c] - mu) * inv * lnw[c] + lnb[c];
    ynorm[((size_t)b * NY + n) * Cdim + c] = (bf16_t)v;
  }
}

// ---------------------------------------------------------------------------
// Generic WMMA GEMM: out[M,N] = A[M,K] @ W[N,K]^T   (A,W bf16, fp32 acc)
// 64x64 block tile, 256 threads = 8 waves, 2 output 16x16 tiles per wave.
// outf != nullptr: fp32 output + bias + residual (final projection).
// ---------------------------------------------------------------------------
__global__ void wmma_gemm_kernel(const bf16_t* __restrict__ A,
                                 const bf16_t* __restrict__ W,
                                 bf16_t* __restrict__ outb,
                                 float* __restrict__ outf,
                                 const float* __restrict__ bias,
                                 const float* __restrict__ resid,
                                 int M, int N, int K) {
  const int m0   = blockIdx.x * 64;
  const int n0   = blockIdx.y * 64;
  const int wave = threadIdx.x >> 5;
  const int lane = threadIdx.x & 31;
  const int ncol  = lane & 15;
  const int mbase = (lane >> 4) * 8;

#pragma unroll
  for (int t = wave * 2; t < wave * 2 + 2; ++t) {
    const int mi = t >> 2, ni = t & 3;
    v8f acc = {};
    for (int kk = 0; kk < K; kk += 32) {
      __builtin_prefetch(A + (size_t)(m0 + mi * 16) * K + kk + 64, 0, 1);
      v16bf af = load_a_frag(A, K, m0 + mi * 16, kk);
      v16bf bf = load_bT_frag(W, K, n0 + ni * 16, kk);
      acc = wmma_bf16(af, bf, acc);
    }
#pragma unroll
    for (int r = 0; r < 8; ++r) {
      const int m = m0 + mi * 16 + mbase + r;
      const int n = n0 + ni * 16 + ncol;
      if (outf) {
        outf[(size_t)m * N + n] = acc[r] + bias[n] + resid[(size_t)m * N + n];
      } else {
        outb[(size_t)m * N + n] = (bf16_t)acc[r];
      }
    }
  }
}

// ---------------------------------------------------------------------------
// Exact top-k threshold: k-th largest of 1024 order-keys held 32-per-lane.
// MSB-first radix select; wave-cooperative count per candidate bit.
// ---------------------------------------------------------------------------
__device__ __forceinline__ uint32_t fkey(float f) {
  uint32_t u = __float_as_uint(f);
  return u ^ ((int)u < 0 ? 0xFFFFFFFFu : 0x80000000u);  // total order
}

__device__ __forceinline__ uint32_t radix_select(const uint32_t* key, int k) {
  uint32_t T = 0;
  for (int bit = 31; bit >= 0; --bit) {
    const uint32_t cand = T | (1u << bit);
    int cnt = 0;
#pragma unroll
    for (int j = 0; j < 32; ++j) cnt += (key[j] >= cand) ? 1 : 0;
#pragma unroll
    for (int off = 16; off; off >>= 1) cnt += __shfl_xor(cnt, off, 32);
    if (cnt >= k) T = cand;
  }
  return T;
}

// ---------------------------------------------------------------------------
// Fused attention: one block = (b, h, 16-query tile).
//   0) async-stage Q tile and the whole V head (1024x64 bf16 = 128K) to LDS;
//      the V stream overlaps phases 1-2 (ASYNCcnt pipelining)
//   1) QK^T via WMMA into 16x1024 fp32 LDS scores (scaled by D^-1/2)
//   2) per-row exact top-k (k1 and k2) + dual masked softmax, fused into ONE
//      combined weight matrix w_i = 0.6*e_i*m1_i/s1 + 0.4*e_i*m2_i/s2 (bf16)
//   3) single P@V WMMA GEMM from LDS-staged V, split-K, LDS reduce, bf16 out
// LDS: 64K scores + 32K probs + 2K q + 4K scratch + 128K V = 230K < 320K/WGP.
// ---------------------------------------------------------------------------
static constexpr int ATTN_SMEM =
    16 * NY * 4 + 16 * NY * 2 + 16 * Dh * 2 + 4 * 16 * 16 * 4 + NY * Dh * 2;

__global__ void attn_topk_kernel(const bf16_t* __restrict__ qm,   // [B,NX,C]
                                 const bf16_t* __restrict__ kvm,  // [B,NY,2C]
                                 bf16_t* __restrict__ attnout,    // [B,NX,C]
                                 const int* __restrict__ k1p,
                                 const int* __restrict__ k2p) {
  extern __shared__ char smem[];
  float*  scores  = (float*)smem;                                  // [16][NY]
  bf16_t* pbuf    = (bf16_t*)(smem + 16 * NY * 4);                 // [16][NY]
  bf16_t* qtile   = (bf16_t*)(smem + 16 * NY * 4 + 16 * NY * 2);   // [16][64]
  float*  scratch = (float*)(smem + 16 * NY * 4 + 16 * NY * 2 + 16 * Dh * 2);
  bf16_t* vtile   = (bf16_t*)(smem + 16 * NY * 4 + 16 * NY * 2 + 16 * Dh * 2 +
                              4 * 16 * 16 * 4);                    // [NY][64]

  const int bid = blockIdx.x;
  const int mt  = bid & 63;           // NX/16 = 64 query tiles
  const int h   = (bid >> 6) % NH;
  const int b   = bid / (64 * NH);
  const int m0  = mt * 16;

  const bf16_t* qbase = qm  + (size_t)b * NX * Cdim + h * Dh;
  const bf16_t* kbase = kvm + (size_t)b * NY * (2 * Cdim) + h * Dh;          // k half
  const bf16_t* vbase = kvm + (size_t)b * NY * (2 * Cdim) + Cdim + h * Dh;   // v half

  const int wave = threadIdx.x >> 5;
  const int lane = threadIdx.x & 31;
  const int ncol  = lane & 15;
  const int mbase = (lane >> 4) * 8;

  // --- phase 0a: async-stage Q tile (16 x 64, 128 x 16B chunks) ---
  for (int i = threadIdx.x; i < (16 * Dh) / 8; i += blockDim.x) {
    const int r = i >> 3, cc = (i & 7) * 8;
    async_g2l_b128(qbase + (size_t)(m0 + r) * Cdim + cc, qtile + r * Dh + cc);
  }
  // --- phase 0b: async-stage V head (1024 x 64, 32 chunks per thread) ---
  for (int i = threadIdx.x; i < (NY * Dh) / 8; i += blockDim.x) {
    const int row = i >> 3, cc = (i & 7) * 8;
    async_g2l_b128(vbase + (size_t)row * (2 * Cdim) + cc, vtile + row * Dh + cc);
  }
  wait_async_le32();   // Q chunks issued first -> retired once <=32 V remain
  __syncthreads();     // publish qtile to all waves

  // --- phase 1: scores = (Q K^T) * D^-1/2  (K streamed from global/L2) ---
  for (int nt = wave; nt < NY / 16; nt += 8) {
    v8f acc = {};
#pragma unroll
    for (int kh = 0; kh < Dh; kh += 32) {
      v16bf af = load_a_frag(qtile, Dh, 0, kh);
      v16bf bf = load_bT_frag(kbase, 2 * Cdim, nt * 16, kh);
      acc = wmma_bf16(af, bf, acc);
    }
#pragma unroll
    for (int r = 0; r < 8; ++r)
      scores[(mbase + r) * NY + nt * 16 + ncol] = acc[r] * 0.125f;  // 64^-0.5
  }
  __syncthreads();

  // --- phase 2: per-row dual top-k masked softmax -> combined weights ---
  const int k1 = *k1p, k2 = *k2p;
  for (int row = wave * 2; row < wave * 2 + 2; ++row) {
    float    sv[32];
    uint32_t key[32];
    float m = -3.4e38f;
#pragma unroll
    for (int j = 0; j < 32; ++j) {
      const float s = scores[row * NY + lane + 32 * j];
      sv[j]  = s;
      key[j] = fkey(s);
      m = fmaxf(m, s);
    }
#pragma unroll
    for (int off = 16; off; off >>= 1) m = fmaxf(m, __shfl_xor(m, off, 32));

    const uint32_t T1 = radix_select(key, k1);   // keep iff key >= T1
    const uint32_t T2 = radix_select(key, k2);   // k2 < k1 -> T2 >= T1

    float e[32];
    float s1 = 0.f, s2 = 0.f;
#pragma unroll
    for (int j = 0; j < 32; ++j) {
      e[j] = __expf(sv[j] - m);
      if (key[j] >= T1) s1 += e[j];
      if (key[j] >= T2) s2 += e[j];
    }
#pragma unroll
    for (int off = 16; off; off >>= 1) {
      s1 += __shfl_xor(s1, off, 32);
      s2 += __shfl_xor(s2, off, 32);
    }
    const float c1 = 0.6f / s1, c2 = 0.4f / s2;
#pragma unroll
    for (int j = 0; j < 32; ++j) {
      float w = 0.f;
      if (key[j] >= T1) w += e[j] * c1;
      if (key[j] >= T2) w += e[j] * c2;
      pbuf[row * NY + lane + 32 * j] = (bf16_t)w;
    }
  }
  wait_async_0();      // all V chunks resident in LDS
  __syncthreads();

  // --- phase 3: out = P @ V from LDS (split-K across wave halves) ---
  const int ntile = wave & 3;      // 4 tiles cover Dh=64
  const int khalf = wave >> 2;
  v8f acc = {};
  for (int kt = khalf * 16; kt < khalf * 16 + 16; ++kt) {
    v16bf af = load_a_frag(pbuf, NY, 0, kt * 32);
    v16bf bf = load_b_rm_frag(vtile, Dh, kt * 32, ntile * 16);
    acc = wmma_bf16(af, bf, acc);
  }
  if (khalf == 1) {
#pragma unroll
    for (int r = 0; r < 8; ++r)
      scratch[ntile * 256 + (mbase + r) * 16 + ncol] = acc[r];
  }
  __syncthreads();
  if (khalf == 0) {
#pragma unroll
    for (int r = 0; r < 8; ++r) {
      const float v = acc[r] + scratch[ntile * 256 + (mbase + r) * 16 + ncol];
      attnout[((size_t)b * NX + m0 + mbase + r) * Cdim + h * Dh + ntile * 16 + ncol] =
          (bf16_t)v;
    }
  }
}

// ---------------------------------------------------------------------------
// Host-side orchestration
// ---------------------------------------------------------------------------
extern "C" void kernel_launch(void* const* d_in, const int* in_sizes, int n_in,
                              void* d_out, int out_size, void* d_ws, size_t ws_size,
                              hipStream_t stream) {
  (void)in_sizes; (void)n_in; (void)out_size; (void)ws_size;

  const float* x      = (const float*)d_in[0];   // [B,NX,C]
  const float* y      = (const float*)d_in[1];   // [B,NY,C]
  const float* c1w    = (const float*)d_in[2];
  const float* c1b    = (const float*)d_in[3];
  const float* c2w    = (const float*)d_in[4];
  const float* c2b    = (const float*)d_in[5];
  const float* c3w    = (const float*)d_in[6];
  const float* c3b    = (const float*)d_in[7];
  const float* lnw    = (const float*)d_in[8];
  const float* lnb    = (const float*)d_in[9];
  const float* q_w    = (const float*)d_in[10];  // [C,C]
  const float* kv_w   = (const float*)d_in[11];  // [2C,C]
  const float* proj_w = (const float*)d_in[12];  // [C,C]
  const float* proj_b = (const float*)d_in[13];
  const int*   k1p    = (const int*)d_in[14];    // device scalar
  const int*   k2p    = (const int*)d_in[15];    // device scalar
  float* out = (float*)d_out;                    // [B,NX,C]

  // workspace carve-up (256B aligned)
  char* ws = (char*)d_ws;
  size_t off = 0;
  auto carve = [&](size_t bytes) {
    char* p = ws + off;
    off = (off + bytes + 255) & ~(size_t)255;
    return p;
  };
  const size_t MX = (size_t)Bsz * NX;            // 4096 rows
  bf16_t* x_bf   = (bf16_t*)carve(MX * Cdim * 2);
  bf16_t* ynorm  = (bf16_t*)carve(MX * Cdim * 2);
  bf16_t* qw_bf  = (bf16_t*)carve((size_t)Cdim * Cdim * 2);
  bf16_t* kvw_bf = (bf16_t*)carve((size_t)2 * Cdim * Cdim * 2);
  bf16_t* pw_bf  = (bf16_t*)carve((size_t)Cdim * Cdim * 2);
  bf16_t* q_bf   = (bf16_t*)carve(MX * Cdim * 2);
  bf16_t* kv_bf  = (bf16_t*)carve(MX * 2 * Cdim * 2);
  bf16_t* ao_bf  = (bf16_t*)carve(MX * Cdim * 2);

  const int THR = 256;
  auto blocks = [&](int n) { return (n + THR - 1) / THR; };

  // 1) casts to bf16
  cast_f32_bf16<<<blocks((int)(MX * Cdim)), THR, 0, stream>>>(x, x_bf, (int)(MX * Cdim));
  cast_f32_bf16<<<blocks(Cdim * Cdim), THR, 0, stream>>>(q_w, qw_bf, Cdim * Cdim);
  cast_f32_bf16<<<blocks(2 * Cdim * Cdim), THR, 0, stream>>>(kv_w, kvw_bf, 2 * Cdim * Cdim);
  cast_f32_bf16<<<blocks(Cdim * Cdim), THR, 0, stream>>>(proj_w, pw_bf, Cdim * Cdim);

  // 2) fused multi-scale depthwise conv + LayerNorm -> ynorm (bf16)
  conv_ln_kernel<<<Bsz * NY, THR, 0, stream>>>(y, c1w, c1b, c2w, c2b, c3w, c3b,
                                               lnw, lnb, ynorm);

  // 3) q = x @ q_w^T            [4096 x 768]
  wmma_gemm_kernel<<<dim3(NX * Bsz / 64, Cdim / 64), THR, 0, stream>>>(
      x_bf, qw_bf, q_bf, nullptr, nullptr, nullptr,
      (int)MX, Cdim, Cdim);

  // 4) kv = ynorm @ kv_w^T      [4096 x 1536]
  wmma_gemm_kernel<<<dim3(NY * Bsz / 64, 2 * Cdim / 64), THR, 0, stream>>>(
      ynorm, kvw_bf, kv_bf, nullptr, nullptr, nullptr,
      (int)MX, 2 * Cdim, Cdim);

  // 5) fused top-k attention (both k1 and k2 paths, single P@V)
  attn_topk_kernel<<<Bsz * NH * (NX / 16), THR, ATTN_SMEM, stream>>>(
      q_bf, kv_bf, ao_bf, k1p, k2p);

  // 6) out = attn_out @ proj_w^T + proj_b + x   (fp32)
  wmma_gemm_kernel<<<dim3(NX * Bsz / 64, Cdim / 64), THR, 0, stream>>>(
      ao_bf, pw_bf, nullptr, out, proj_b, x,
      (int)MX, Cdim, Cdim);
}
